// MI_loss_17334488007391
// MI455X (gfx1250) — matrix operand order repaired and checked by reading, not compiled, not run on hardware
//
#include <hip/hip_runtime.h>
#include <hip/hip_bf16.h>
#include <math.h>

typedef __attribute__((ext_vector_type(16))) _Float16 v16h;
typedef __attribute__((ext_vector_type(4)))  _Float16 v4h;
typedef __attribute__((ext_vector_type(8)))  float    v8f;
typedef __attribute__((ext_vector_type(4)))  float    f32x4;

#define NUM_TASKS   8
#define NUM_EXPERTS 8
// acc layout: [task][col], col 0..7 = seg_sum over experts, col 8 = count
#define ACC_COLS    (NUM_EXPERTS + 1)
#define ACC_ELEMS   (NUM_TASKS * ACC_COLS)

// Padded per-token row stride (in halves) for the staged B tile.
// 20 halves = 40 B: 8 B-aligned lane bases (b64-mergeable stores) and the
// two lane-group gather windows hit disjoint LDS banks (160 dwords = 32 mod 64).
#define PSTR 20

// ---------------------------------------------------------------------------
// Zero the 72-float global accumulator (ws scratch).
// ---------------------------------------------------------------------------
__global__ void mi_init_kernel(float* __restrict__ acc) {
    int i = threadIdx.x;
    if (i < ACC_ELEMS) acc[i] = 0.0f;
}

// ---------------------------------------------------------------------------
// Main pass: softmax per token, then segment-sum via WMMA:
//   D(16x16,f32) += A(16x32 one-hot of labels, f16) x B(32x16 [probs|ones], f16)
// Each wave processes 32 tokens per WMMA, accumulating its private C tile.
// Tail: wave tiles -> block LDS accumulator (ds_add_f32) -> global atomics.
// ---------------------------------------------------------------------------
__global__ __launch_bounds__(256) void mi_main_kernel(
        const float* __restrict__ logits,   // [N, 8] row-major
        const int*   __restrict__ labels,   // [N]
        float*       __restrict__ acc,      // [8][9] global accumulator
        int n_tokens, int n_tiles)
{
    // Per-wave private staging slices (8 waves per 256-thread block).
    __shared__ _Float16 sProb[8][32 * PSTR];  // B tile staging: [tokenK][col]
    __shared__ int      sLab [8][32];         // labels of the wave's 32 tokens
    __shared__ float    sAcc [ACC_ELEMS];     // block-level partial reduction

    const int lane  = threadIdx.x & 31;
    const int wav   = threadIdx.x >> 5;
    const int group = lane >> 4;     // lane half: selects K-range of fragment
    const int idx   = lane & 15;     // A: row m; B: col n; C: col n

    const int waves_total = gridDim.x * (blockDim.x >> 5);
    const int wave_global = blockIdx.x * (blockDim.x >> 5) + wav;

    // Columns 9..15 of the B tile are never rewritten: zero once.
    #pragma unroll
    for (int h = 9; h < 16; ++h)
        sProb[wav][lane * PSTR + h] = (_Float16)0.0f;
    if (threadIdx.x < ACC_ELEMS) sAcc[threadIdx.x] = 0.0f;
    __syncthreads();

    v8f c = {};  // f32 accumulator tile: rows = tasks, cols = experts|count

    // Uniform trip count across the whole grid -> EXEC all-1s at every WMMA
    // and __syncthreads() is always convergent.
    const int iters = (n_tiles + waves_total - 1) / waves_total;

    for (int it = 0; it < iters; ++it) {
        const int tile  = wave_global + it * waves_total;
        const int token = tile * 32 + lane;
        const bool valid = (token < n_tokens) && (tile < n_tiles);
        const int  t     = valid ? token : 0;

        // ---- per-lane softmax over 8 experts (two NT b128 loads) ----
        const f32x4* lp = (const f32x4*)(logits + (size_t)t * NUM_EXPERTS);
        f32x4 lo = __builtin_nontemporal_load(lp);
        f32x4 hi = __builtin_nontemporal_load(lp + 1);
        float x[8] = { lo.x, lo.y, lo.z, lo.w, hi.x, hi.y, hi.z, hi.w };

        float mx = x[0];
        #pragma unroll
        for (int e = 1; e < 8; ++e) mx = fmaxf(mx, x[e]);
        float ex[8], s = 0.0f;
        #pragma unroll
        for (int e = 0; e < 8; ++e) { ex[e] = __expf(x[e] - mx); s += ex[e]; }
        const float r = 1.0f / s;

        // ---- stage this token's B-tile row (probs + ones col) and label ----
        v4h p0, p1;
        #pragma unroll
        for (int e = 0; e < 4; ++e) p0[e] = (_Float16)(ex[e] * r);
        #pragma unroll
        for (int e = 0; e < 4; ++e) p1[e] = (_Float16)(ex[e + 4] * r);
        v4h* prow = (v4h*)&sProb[wav][lane * PSTR];  // 8 B-aligned (PSTR=20)
        prow[0] = p0;                                // two ds_store_b64
        prow[1] = p1;
        sProb[wav][lane * PSTR + 8] = valid ? (_Float16)1.0f : (_Float16)0.0f;
        sLab[wav][lane] = valid ? __builtin_nontemporal_load(labels + t) : 100;

        __syncthreads();

        // ---- gather WMMA fragments, contiguous K order (same for A and B) ----
        // A (16x32, MxK): lane row m = idx; group0 K = 0..15,
        //                 group1 K = 16..31
        v16h a;
        #pragma unroll
        for (int h = 0; h < 16; ++h) {
            const int K = h + group * 16;
            a[h] = (sLab[wav][K] == idx) ? (_Float16)1.0f : (_Float16)0.0f;
        }
        // B (32x16, KxN): lane col n = idx; group0 K = 0..15, group1 K = 16..31
        v16h b;
        #pragma unroll
        for (int h = 0; h < 16; ++h) {
            const int K = h + group * 16;
            b[h] = sProb[wav][K * PSTR + idx];
        }

        // D = A x B + C  (f32 accumulate)
        c = __builtin_amdgcn_wmma_f32_16x16x32_f16(
                /*neg_a=*/false, a, /*neg_b=*/false, b,
                /*c_mod=*/(short)0, c, /*reuse_a=*/false, /*reuse_b=*/false);
    }

    // C layout (32-bit 16x16): lanes 0-15 hold rows 0-7 in v[0..7], col = lane.
    // Live corner is rows 0..7 (tasks), cols 0..8 (experts + count).
    // Stage 1: fold each wave's tile into the block accumulator (ds_add_f32).
    if (group == 0 && idx < ACC_COLS) {
        #pragma unroll
        for (int m = 0; m < NUM_TASKS; ++m)
            atomicAdd(&sAcc[m * ACC_COLS + idx], c[m]);
    }
    __syncthreads();

    // Stage 2: one global atomic per block per element.
    if (threadIdx.x < ACC_ELEMS)
        atomicAdd(&acc[threadIdx.x], sAcc[threadIdx.x]);
}

// ---------------------------------------------------------------------------
// Finalize: 8x9 accumulator -> mutual-information loss scalar.
// ---------------------------------------------------------------------------
__global__ void mi_final_kernel(const float* __restrict__ acc,
                                float* __restrict__ out)
{
    if (threadIdx.x != 0 || blockIdx.x != 0) return;

    float EX[NUM_TASKS][NUM_EXPERTS];
    float tot = 0.0f;
    for (int tk = 0; tk < NUM_TASKS; ++tk) {
        const float cnt = acc[tk * ACC_COLS + NUM_EXPERTS];
        for (int e = 0; e < NUM_EXPERTS; ++e) {
            const float v = acc[tk * ACC_COLS + e] * cnt;
            EX[tk][e] = v;
            tot += v;
        }
    }
    tot *= 0.5f;                       // / TOPK (=2)
    const float inv = 1.0f / (tot + 1e-4f);

    float P_TI[NUM_TASKS], P_EI[NUM_EXPERTS];
    for (int e = 0; e < NUM_EXPERTS; ++e) P_EI[e] = 0.0f;
    for (int tk = 0; tk < NUM_TASKS; ++tk) {
        float rs = 0.0f;
        for (int e = 0; e < NUM_EXPERTS; ++e) {
            EX[tk][e] *= inv;
            rs += EX[tk][e];
            P_EI[e] += EX[tk][e];
        }
        P_TI[tk] = rs + 1e-4f;
    }
    for (int e = 0; e < NUM_EXPERTS; ++e) P_EI[e] += 1e-4f;

    float loss = 0.0f;
    for (int tk = 0; tk < NUM_TASKS; ++tk)
        for (int e = 0; e < NUM_EXPERTS; ++e)
            loss -= EX[tk][e] * logf(EX[tk][e] / (P_TI[tk] * P_EI[e]) + 1e-4f);

    out[0] = 0.01f * loss;             // WEX * expert_loss
}

// ---------------------------------------------------------------------------
extern "C" void kernel_launch(void* const* d_in, const int* in_sizes, int n_in,
                              void* d_out, int out_size, void* d_ws, size_t ws_size,
                              hipStream_t stream) {
    const float* logits = (const float*)d_in[0];
    const int*   labels = (const int*)d_in[1];
    float*       out    = (float*)d_out;
    float*       acc    = (float*)d_ws;      // 72 floats of scratch

    const int n_tokens = in_sizes[1];        // labels element count == N
    const int n_tiles  = (n_tokens + 31) / 32;

    mi_init_kernel<<<1, 128, 0, stream>>>(acc);

    const int blocks  = 512;                 // 4096 waves -> ~32 tiles/wave
    const int threads = 256;                 // 8 wave32 per block
    mi_main_kernel<<<blocks, threads, 0, stream>>>(logits, labels, acc,
                                                   n_tokens, n_tiles);

    mi_final_kernel<<<1, 32, 0, stream>>>(acc, out);
}
